// CBSModel_31585189495113
// MI455X (gfx1250) — matrix operand order repaired and checked by reading, not compiled, not run on hardware
//
#include <hip/hip_runtime.h>
#include <hip/hip_bf16.h>

// ---------------------------------------------------------------------------
// CBS trajectory model for MI455X (gfx1250, wave32, WMMA).
// All large GEMMs (LSTM gates, projections, MLPs) run through one bf16-WMMA
// GEMM kernel (fp32 accumulate). Weights are converted to bf16 ONCE per
// launch (reused 30-50x by the recurrences), so the GEMM inner loop has no
// W-side conversions and A-fragments are two ds_load_b128 each.
// Primitives (P=4) / modes (M=6) are batched into gridDim.z so every
// sequential LSTM step is a single wide GEMM.
// GEMM block tile: 64x64, 4 waves, each wave = 16-wide N strip x 64 M rows
// (4 accumulators, B-fragment reused across 4 WMMAs).
// Requires: Mrows % 64 == 0, N % 64 == 0, K % 32 == 0 (true for all calls).
// ---------------------------------------------------------------------------

namespace {

constexpr int Bn  = 1024;  // batch
constexpr int TH  = 50;    // history length
constexpr int TF  = 30;    // future length
constexpr int DIN = 8;
constexpr int H   = 256;
constexpr int Mm  = 6;     // modes
constexpr int Pp  = 4;     // primitives
constexpr int NHh = 4;     // attention heads
constexpr int HD  = 64;    // head dim

typedef __attribute__((ext_vector_type(16))) __bf16 v16bf;
typedef __attribute__((ext_vector_type(8)))  __bf16 v8bf;
typedef __attribute__((ext_vector_type(4)))  __bf16 v4bf;
typedef __attribute__((ext_vector_type(8)))  float  v8f;

__device__ __forceinline__ float sigm(float x) { return 1.f / (1.f + __expf(-x)); }

// ---------------------------------------------------------------------------
// WMMA GEMM:  C[z] = A0[z] @ W0[z]^T + A1[z] @ W1[z]^T + bias0[z] + bias1[z]
//   A: (Mrows x K) f32 row-major, leading dim ldA (per-z stride aB)
//   W: (N x K) bf16 row-major (pre-converted)     (per-z stride wB)
//   C: (Mrows x N) f32 row-major                  (per-z stride cB)
// blockDim = 128 (4 waves). grid = (N/64, Mrows/64, nbatch).
// ---------------------------------------------------------------------------
__global__ __launch_bounds__(128)
void gemm_wmma_bf16(const float*  __restrict__ A0, long aB0, int ldA0,
                    const __bf16* __restrict__ W0, long wB0,
                    const float*  __restrict__ bias0, long bB0, int K0,
                    const float*  __restrict__ A1, long aB1, int ldA1,
                    const __bf16* __restrict__ W1, long wB1,
                    const float*  __restrict__ bias1, long bB1, int K1,
                    float* __restrict__ C, long cB,
                    int Mrows, int N, int relu)
{
    // 40-bf16 (80B) row stride: rows stay 16B-aligned for b128 LDS ops, and
    // the 16 parallel per-fragment b128 loads (start banks 20*r mod 64, all
    // multiples of 4, 16 loads x 4 banks = 64 banks) are conflict-free.
    __shared__ __bf16 Asb[64][40];   // 64 M rows x 32 K, A stored bf16
    __shared__ __bf16 Wsb[64][40];   // 64 N rows x 32 K, W copied bf16

    const int lane = threadIdx.x & 31;
    const int wave = threadIdx.x >> 5;
    const int half = lane >> 4;
    const int l15  = lane & 15;
    const long nBase = (long)blockIdx.x * 64;
    const long mBase = (long)blockIdx.y * 64;
    const int z = blockIdx.z;

    v8f acc[4];
    #pragma unroll
    for (int mt = 0; mt < 4; ++mt)
        acc[mt] = (v8f){0.f, 0.f, 0.f, 0.f, 0.f, 0.f, 0.f, 0.f};

    for (int s = 0; s < 2; ++s) {
        const float*  A  = s ? A1  : A0;
        const __bf16* Wp = s ? W1  : W0;
        const int     K  = s ? K1  : K0;
        const int     ld = s ? ldA1 : ldA0;
        const long    aB = s ? aB1 : aB0;
        const long    wB = s ? wB1 : wB0;
        if (A == nullptr || Wp == nullptr || K <= 0) continue;
        A  += (long)z * aB;
        Wp += (long)z * wB;

        for (int kc = 0; kc < K; kc += 32) {
            // A tile: 64x32 f32 -> bf16 in WMMA K-order (contiguous K runs)
            #pragma unroll
            for (int i = threadIdx.x; i < 64 * 8; i += 128) {   // 512 float4
                int r = i >> 3, q = (i & 7) * 4;
                float4 va = *(const float4*)&A[(mBase + r) * (long)ld + kc + q];
                v4bf vb;
                vb[0] = (__bf16)va.x; vb[1] = (__bf16)va.y;
                vb[2] = (__bf16)va.z; vb[3] = (__bf16)va.w;
                *(v4bf*)&Asb[r][q] = vb;
            }
            // W tile: straight bf16 copy, 16B per iteration
            #pragma unroll
            for (int i = threadIdx.x; i < 64 * 4; i += 128) {   // 256 x 8 bf16
                int r = i >> 2, q = (i & 3) * 8;
                *(v8bf*)&Wsb[r][q] =
                    *(const v8bf*)&Wp[(nBase + r) * (long)K + kc + q];
            }
            // prefetch next K chunk (global_prefetch_b8; semantically inert)
            if (kc + 32 < K) {
                int r = threadIdx.x >> 1;
                __builtin_prefetch(&A[(mBase + r) * (long)ld + kc + 32], 0, 1);
                __builtin_prefetch(&Wp[(nBase + r) * (long)K + kc + 32], 0, 1);
            }
            __syncthreads();

            // B fragment (K x N): lane -> n, element j -> k (same K order as A)
            v16bf b;
            #pragma unroll
            for (int j = 0; j < 16; ++j)
                b[j] = Wsb[wave * 16 + l15][half * 8 + (j & 7) + (j >> 3) * 16];

            // 4 A fragments (one per 16-row M sub-tile), B reused.
            // A frag: lane -> m; element 2r+e holds
            // K = (r/4)*16 + half*8 + (r%4)*2 + e  (ISA 7.12.2), i.e. two
            // contiguous 8-element runs -> two 16B LDS loads, zero converts.
            #pragma unroll
            for (int mt = 0; mt < 4; ++mt) {
                const __bf16* arow = &Asb[mt * 16 + l15][0];
                v8bf lo = *(const v8bf*)(arow + half * 8);
                v8bf hi = *(const v8bf*)(arow + 16 + half * 8);
                v16bf a = __builtin_shufflevector(
                    lo, hi, 0, 1, 2, 3, 4, 5, 6, 7,
                    8, 9, 10, 11, 12, 13, 14, 15);
                acc[mt] = __builtin_amdgcn_wmma_f32_16x16x32_bf16(
                    false, a, false, b, (short)0, acc[mt], false, false);
            }
            __syncthreads();
        }
    }

    const float* b0 = bias0 ? bias0 + (long)z * bB0 : nullptr;
    const float* b1 = bias1 ? bias1 + (long)z * bB1 : nullptr;
    float* Cz = C + (long)z * cB;
    const long n = nBase + wave * 16 + l15;
    const float badd = (b0 ? b0[n] : 0.f) + (b1 ? b1[n] : 0.f);
    #pragma unroll
    for (int mt = 0; mt < 4; ++mt) {
        #pragma unroll
        for (int v = 0; v < 8; ++v) {
            long m = mBase + mt * 16 + v + 8 * half;  // C/D layout (ISA 7.12.2)
            float val = acc[mt][v] + badd;
            if (relu) val = fmaxf(val, 0.f);
            Cz[m * (long)N + n] = val;
        }
    }
}

// ------------------------------- elementwise -------------------------------

__global__ void zero_kernel(float* p, long n) {
    long i = (long)blockIdx.x * blockDim.x + threadIdx.x;
    for (; i < n; i += (long)gridDim.x * blockDim.x) p[i] = 0.f;
}

// f32 -> bf16 weight conversion (n % 4 == 0), 4 elements per thread
__global__ void cvt_bf16_kernel(const float* __restrict__ s,
                                __bf16* __restrict__ d, long n) {
    long i = ((long)blockIdx.x * blockDim.x + threadIdx.x) * 4;
    if (i >= n) return;
    float4 v = *(const float4*)&s[i];
    v4bf o;
    o[0] = (__bf16)v.x; o[1] = (__bf16)v.y; o[2] = (__bf16)v.z; o[3] = (__bf16)v.w;
    *(v4bf*)&d[i] = o;
}

// x = relu(history @ W_in^T + b_in)   threads: B*TH*H
__global__ void input_proj_kernel(const float* __restrict__ hist,
                                  const float* __restrict__ Win,
                                  const float* __restrict__ bin,
                                  float* __restrict__ x) {
    long idx = (long)blockIdx.x * blockDim.x + threadIdx.x;
    if (idx >= (long)Bn * TH * H) return;
    long row = idx / H; int j = (int)(idx % H);
    float s = bin[j];
    #pragma unroll
    for (int d = 0; d < DIN; ++d) s += hist[row * DIN + d] * Win[j * DIN + d];
    x[idx] = fmaxf(s, 0.f);
}

// LSTM cell pointwise: gates (rows x 4H) -> update h,c (rows x H)
__global__ void lstm_cell_kernel(const float* __restrict__ gates,
                                 float* __restrict__ h, float* __restrict__ c,
                                 int rows) {
    long idx = (long)blockIdx.x * blockDim.x + threadIdx.x;
    if (idx >= (long)rows * H) return;
    long row = idx / H; int j = (int)(idx % H);
    const float* g = gates + row * 4L * H;
    float ig = sigm(g[j]);
    float fg = sigm(g[H + j]);
    float gg = tanhf(g[2 * H + j]);
    float og = sigm(g[3 * H + j]);
    float cn = fg * c[idx] + ig * gg;
    c[idx] = cn;
    h[idx] = og * tanhf(cn);
}

// k = prim_emb @ Wk^T + bk     threads: P*H
__global__ void kproj_kernel(const float* __restrict__ pe,
                             const float* __restrict__ Wk,
                             const float* __restrict__ bk,
                             float* __restrict__ kb) {
    int idx = blockIdx.x * blockDim.x + threadIdx.x;
    if (idx >= Pp * H) return;
    int p = idx / H, j = idx % H;
    float s = bk[j];
    for (int r = 0; r < H; ++r) s += pe[p * H + r] * Wk[j * H + r];
    kb[idx] = s;
}

// MHA over P=4 keys.   threads: B*NH*P  (one per (b, head, query-primitive))
__global__ void attention_kernel(const float* __restrict__ q,
                                 const float* __restrict__ kb,
                                 const float* __restrict__ v,
                                 float* __restrict__ att) {
    int idx = blockIdx.x * blockDim.x + threadIdx.x;
    if (idx >= Bn * NHh * Pp) return;
    int qp = idx % Pp;
    int hh = (idx / Pp) % NHh;
    int b  = idx / (Pp * NHh);
    const float* qr = q + ((long)b * Pp + qp) * H + hh * HD;
    float sc[Pp]; float mx = -1e30f;
    for (int kp = 0; kp < Pp; ++kp) {
        const float* kr = kb + kp * H + hh * HD;
        float s = 0.f;
        for (int d = 0; d < HD; ++d) s += qr[d] * kr[d];
        sc[kp] = s * 0.125f;                  // hd^-0.5 = 1/8
        mx = fmaxf(mx, sc[kp]);
    }
    float den = 0.f;
    for (int kp = 0; kp < Pp; ++kp) { sc[kp] = __expf(sc[kp] - mx); den += sc[kp]; }
    for (int d = 0; d < HD; ++d) {
        float acc = 0.f;
        for (int kp = 0; kp < Pp; ++kp)
            acc += sc[kp] * v[((long)b * Pp + kp) * H + hh * HD + d];
        att[((long)b * Pp + qp) * H + hh * HD + d] = acc / den;
    }
}

// composed += mean_p attO[b,p,:]    threads: B*H
__global__ void agg_add_kernel(const float* __restrict__ attO,
                               float* __restrict__ composed) {
    long idx = (long)blockIdx.x * blockDim.x + threadIdx.x;
    if (idx >= (long)Bn * H) return;
    long b = idx / H; int j = (int)(idx % H);
    float s = 0.f;
    for (int p = 0; p < Pp; ++p) s += attO[(b * Pp + p) * (long)H + j];
    composed[idx] += s * (1.f / Pp);
}

// encP (P,B,H) -> encT (B,P,H)
__global__ void transpose_pb_kernel(const float* __restrict__ encP,
                                    float* __restrict__ encT) {
    long idx = (long)blockIdx.x * blockDim.x + threadIdx.x;
    if (idx >= (long)Bn * Pp * H) return;
    long b = idx / ((long)Pp * H);
    int p = (int)((idx / H) % Pp);
    int j = (int)(idx % H);
    encT[idx] = encP[((long)p * Bn + b) * H + j];
}

// te[t,:] = relu(tn*Wt1 + bt1) @ Wt2^T + bt2     threads: TF*H
__global__ void time_emb_kernel(const float* __restrict__ Wt1,
                                const float* __restrict__ bt1,
                                const float* __restrict__ Wt2,
                                const float* __restrict__ bt2,
                                float* __restrict__ te) {
    int idx = blockIdx.x * blockDim.x + threadIdx.x;
    if (idx >= TF * H) return;
    int t = idx / H, j = idx % H;
    float tn = (float)t / (float)(TF - 1);
    float acc = bt2[j];
    for (int r = 0; r < H / 2; ++r)
        acc += fmaxf(tn * Wt1[r] + bt1[r], 0.f) * Wt2[j * (H / 2) + r];
    te[idx] = acc;
}

// split init (M,B,2H) -> h0,c0 (and replicate into h1,c1)    threads: M*B*H
__global__ void split_init_kernel(const float* __restrict__ ini,
                                  float* __restrict__ h0, float* __restrict__ c0,
                                  float* __restrict__ h1, float* __restrict__ c1) {
    long idx = (long)blockIdx.x * blockDim.x + threadIdx.x;
    if (idx >= (long)Mm * Bn * H) return;
    long mb = idx / H; int j = (int)(idx % H);
    float v0 = ini[mb * 2L * H + j];
    float v1 = ini[mb * 2L * H + H + j];
    h0[idx] = v0; c0[idx] = v1; h1[idx] = v0; c1[idx] = v1;
}

// xin = mf + 0.5*te[t] + 0.3*concat(prev@Wf^T + bf, 0)    threads: M*B*H
__global__ void xin_kernel(const float* __restrict__ mf,
                           const float* __restrict__ te,
                           const float* __restrict__ prev,
                           const float* __restrict__ Wf,
                           const float* __restrict__ bf,
                           float* __restrict__ xin, int t) {
    long idx = (long)blockIdx.x * blockDim.x + threadIdx.x;
    if (idx >= (long)Mm * Bn * H) return;
    long mb = idx / H; int j = (int)(idx % H);
    int m = (int)(mb / Bn);
    float f = mf[idx] + 0.5f * te[t * H + j];
    if (j < H / 2) {
        float p0 = prev[mb * 2], p1 = prev[mb * 2 + 1];
        long wo = (long)m * (H / 2) * 2 + (long)j * 2;
        float fb = p0 * Wf[wo] + p1 * Wf[wo + 1] + bf[m * (H / 2) + j];
        f += 0.3f * fb;
    }
    xin[idx] = f;
}

// out = o1 @ Wo2^T + bo2 ; prev = out ; trajs[b,m,t,:] = (t==0 ? 0 : out)
__global__ void out_head_kernel(const float* __restrict__ o1,
                                const float* __restrict__ Wo2,
                                const float* __restrict__ bo2,
                                float* __restrict__ prev,
                                float* __restrict__ trajs, int t) {
    int idx = blockIdx.x * blockDim.x + threadIdx.x;
    if (idx >= Mm * Bn) return;
    int m = idx / Bn, b = idx % Bn;
    const float* row = o1 + (long)idx * H;
    #pragma unroll
    for (int n = 0; n < 2; ++n) {
        const float* w = Wo2 + ((long)m * 2 + n) * H;
        float s = bo2[m * 2 + n];
        for (int r = 0; r < H; ++r) s += row[r] * w[r];
        prev[(long)idx * 2 + n] = s;
        trajs[(((long)b * Mm + m) * TF + t) * 2 + n] = (t == 0) ? 0.f : s;
    }
}

// logits = chid @ W2^T + b2 ; conf = softmax(logits/1.2)    threads: B
__global__ void conf_kernel(const float* __restrict__ chid,
                            const float* __restrict__ W2,
                            const float* __restrict__ b2,
                            float* __restrict__ outConf,
                            float* __restrict__ outLogits) {
    int b = blockIdx.x * blockDim.x + threadIdx.x;
    if (b >= Bn) return;
    float lg[Mm];
    for (int n = 0; n < Mm; ++n) {
        float s = b2[n];
        for (int r = 0; r < H / 2; ++r) s += chid[(long)b * (H / 2) + r] * W2[n * (H / 2) + r];
        lg[n] = s;
        outLogits[(long)b * Mm + n] = s;
    }
    float mx = -1e30f;
    for (int n = 0; n < Mm; ++n) mx = fmaxf(mx, lg[n] / 1.2f);
    float den = 0.f, e[Mm];
    for (int n = 0; n < Mm; ++n) { e[n] = __expf(lg[n] / 1.2f - mx); den += e[n]; }
    for (int n = 0; n < Mm; ++n) outConf[(long)b * Mm + n] = e[n] / den;
}

// -------------------------------- host side --------------------------------

static inline void gemm(hipStream_t s,
    const float* A0, long aB0, int ldA0, const __bf16* W0, long wB0,
    const float* b0, long bB0, int K0,
    const float* A1, long aB1, int ldA1, const __bf16* W1, long wB1,
    const float* b1, long bB1, int K1,
    float* C, long cB, int Mrows, int N, int nb, int relu)
{
    dim3 g(N / 64, Mrows / 64, nb);
    gemm_wmma_bf16<<<g, 128, 0, s>>>(A0, aB0, ldA0, W0, wB0, b0, bB0, K0,
                                     A1, aB1, ldA1, W1, wB1, b1, bB1, K1,
                                     C, cB, Mrows, N, relu);
}

static inline dim3 ew(long n) { return dim3((unsigned)((n + 255) / 256)); }

} // namespace

extern "C" void kernel_launch(void* const* d_in, const int* in_sizes, int n_in,
                              void* d_out, int out_size, void* d_ws, size_t ws_size,
                              hipStream_t stream) {
    (void)in_sizes; (void)n_in; (void)out_size; (void)ws_size;
    int i = 0;
    auto F = [&]() { return (const float*)d_in[i++]; };

    const float* history = F();
    const float* W_in = F();  const float* b_in = F();
    // prim
    const float* Wih0 = F(); const float* Whh0 = F(); const float* bih0 = F(); const float* bhh0 = F();
    const float* Wih1 = F(); const float* Whh1 = F(); const float* bih1 = F(); const float* bhh1 = F();
    const float* Wp1 = F();  const float* bp1 = F();  const float* Wp2 = F();  const float* bp2 = F();
    // comp
    const float* prim_emb = F();
    const float* Wq = F(); const float* bq = F();
    const float* Wk = F(); const float* bk = F();
    const float* Wv = F(); const float* bv = F();
    const float* Wo = F(); const float* bo = F();
    const float* Wc1 = F(); const float* bc1 = F();
    const float* Wc2 = F(); const float* bc2 = F();
    // time
    const float* Wt1 = F(); const float* bt1 = F();
    const float* Wt2 = F(); const float* bt2 = F();
    // modes
    const float* Wm1 = F(); const float* bm1 = F();
    const float* Wm2 = F(); const float* bm2 = F();
    const float* Wm3 = F(); const float* bm3 = F();
    const float* Winit = F(); const float* binit = F();
    const float* Wf = F(); const float* bf = F();
    const float* dWih0 = F(); const float* dWhh0 = F(); const float* dbih0 = F(); const float* dbhh0 = F();
    const float* dWih1 = F(); const float* dWhh1 = F(); const float* dbih1 = F(); const float* dbhh1 = F();
    const float* Wo1 = F(); const float* bo1 = F();
    const float* Wo2 = F(); const float* bo2 = F();
    // conf
    const float* cW1 = F(); const float* cb1 = F();
    const float* cW2 = F(); const float* cb2 = F();

    // ---- workspace layout (f32 buffers) ----
    float* ws = (float*)d_ws;
    size_t off = 0;
    auto alloc = [&](size_t n) { float* p = ws + off; off += n; return p; };
    float* x     = alloc((size_t)Bn * TH * H);      // 13.1M floats
    float* h0e   = alloc((size_t)Pp * Bn * H);
    float* c0e   = alloc((size_t)Pp * Bn * H);
    float* h1e   = alloc((size_t)Pp * Bn * H);
    float* c1e   = alloc((size_t)Pp * Bn * H);
    float* gatesE= alloc((size_t)Pp * Bn * 4 * H);
    float* zbuf  = alloc((size_t)Pp * Bn * H);
    float* encP  = alloc((size_t)Pp * Bn * H);
    float* encT  = alloc((size_t)Bn * Pp * H);
    float* kbuf  = alloc((size_t)Pp * H);
    float* qbuf  = alloc((size_t)Bn * Pp * H);
    float* vbuf  = alloc((size_t)Bn * Pp * H);
    float* attb  = alloc((size_t)Bn * Pp * H);
    float* attO  = alloc((size_t)Bn * Pp * H);
    float* comp1 = alloc((size_t)Bn * 2 * H);
    float* composed = alloc((size_t)Bn * H);
    float* te    = alloc((size_t)TF * H);
    float* mf1   = alloc((size_t)Mm * Bn * 2 * H);
    float* mf2b  = alloc((size_t)Mm * Bn * H);
    float* mfb   = alloc((size_t)Mm * Bn * H);
    float* initb = alloc((size_t)Mm * Bn * 2 * H);
    float* dh0   = alloc((size_t)Mm * Bn * H);
    float* dc0   = alloc((size_t)Mm * Bn * H);
    float* dh1   = alloc((size_t)Mm * Bn * H);
    float* dc1   = alloc((size_t)Mm * Bn * H);
    float* dgates= alloc((size_t)Mm * Bn * 4 * H);
    float* xinb  = alloc((size_t)Mm * Bn * H);
    float* o1b   = alloc((size_t)Mm * Bn * H);
    float* prevb = alloc((size_t)Mm * Bn * 2);
    float* chid  = alloc((size_t)Bn * (H / 2));

    // ---- bf16 weight arena (converted once per launch) ----
    off = (off + 3) & ~(size_t)3;                      // 16B align
    __bf16* bfp = (__bf16*)(ws + off);
    auto CV = [&](const float* src, size_t n) {
        __bf16* dst = bfp; bfp += n;
        cvt_bf16_kernel<<<ew((long)(n / 4)), 256, 0, stream>>>(src, dst, (long)n);
        return (const __bf16*)dst;
    };
    const __bf16* bWih0 = CV(Wih0, (size_t)Pp * 4 * H * H);
    const __bf16* bWhh0 = CV(Whh0, (size_t)Pp * 4 * H * H);
    const __bf16* bWih1 = CV(Wih1, (size_t)Pp * 4 * H * H);
    const __bf16* bWhh1 = CV(Whh1, (size_t)Pp * 4 * H * H);
    const __bf16* bWp1  = CV(Wp1,  (size_t)Pp * H * H);
    const __bf16* bWp2  = CV(Wp2,  (size_t)Pp * H * H);
    const __bf16* bWq   = CV(Wq,   (size_t)H * H);
    const __bf16* bWv   = CV(Wv,   (size_t)H * H);
    const __bf16* bWo   = CV(Wo,   (size_t)H * H);
    const __bf16* bWc1  = CV(Wc1,  (size_t)2 * H * Pp * H);
    const __bf16* bWc2  = CV(Wc2,  (size_t)H * 2 * H);
    const __bf16* bWm1  = CV(Wm1,  (size_t)Mm * 2 * H * H);
    const __bf16* bWm2  = CV(Wm2,  (size_t)Mm * H * 2 * H);
    const __bf16* bWm3  = CV(Wm3,  (size_t)Mm * H * H);
    const __bf16* bWinit= CV(Winit,(size_t)Mm * 2 * H * H);
    const __bf16* bdWih0= CV(dWih0,(size_t)Mm * 4 * H * H);
    const __bf16* bdWhh0= CV(dWhh0,(size_t)Mm * 4 * H * H);
    const __bf16* bdWih1= CV(dWih1,(size_t)Mm * 4 * H * H);
    const __bf16* bdWhh1= CV(dWhh1,(size_t)Mm * 4 * H * H);
    const __bf16* bWo1  = CV(Wo1,  (size_t)Mm * H * H);
    const __bf16* bcW1  = CV(cW1,  (size_t)(H / 2) * H);

    float* outTrajs  = (float*)d_out;                          // (B,M,TF,2)
    float* outConf   = outTrajs + (size_t)Bn * Mm * TF * 2;    // (B,M)
    float* outLogits = outConf + (size_t)Bn * Mm;              // (B,M)

    // ---- zero recurrent state ----
    zero_kernel<<<dim3(2048), 256, 0, stream>>>(h0e, (long)4 * Pp * Bn * H); // h0e..c1e contiguous
    zero_kernel<<<dim3(64), 256, 0, stream>>>(prevb, (long)Mm * Bn * 2);

    // ---- input projection ----
    input_proj_kernel<<<ew((long)Bn * TH * H), 256, 0, stream>>>(history, W_in, b_in, x);

    // ---- encoder: 2-layer LSTM, all P primitives batched via grid.z ----
    for (int t = 0; t < TH; ++t) {
        gemm(stream,
             x + (long)t * H, 0, TH * H, bWih0, 4L * H * H, bih0, 4L * H, H,
             h0e, (long)Bn * H, H,       bWhh0, 4L * H * H, bhh0, 4L * H, H,
             gatesE, (long)Bn * 4 * H, Bn, 4 * H, Pp, 0);
        lstm_cell_kernel<<<ew((long)Pp * Bn * H), 256, 0, stream>>>(gatesE, h0e, c0e, Pp * Bn);
        gemm(stream,
             h0e, (long)Bn * H, H, bWih1, 4L * H * H, bih1, 4L * H, H,
             h1e, (long)Bn * H, H, bWhh1, 4L * H * H, bhh1, 4L * H, H,
             gatesE, (long)Bn * 4 * H, Bn, 4 * H, Pp, 0);
        lstm_cell_kernel<<<ew((long)Pp * Bn * H), 256, 0, stream>>>(gatesE, h1e, c1e, Pp * Bn);
    }
    // z = relu(h1 @ Wp1^T + bp1); enc = z @ Wp2^T + bp2
    gemm(stream, h1e, (long)Bn * H, H, bWp1, (long)H * H, bp1, (long)H, H,
         nullptr, 0, 0, nullptr, 0, nullptr, 0, 0,
         zbuf, (long)Bn * H, Bn, H, Pp, 1);
    gemm(stream, zbuf, (long)Bn * H, H, bWp2, (long)H * H, bp2, (long)H, H,
         nullptr, 0, 0, nullptr, 0, nullptr, 0, 0,
         encP, (long)Bn * H, Bn, H, Pp, 0);
    transpose_pb_kernel<<<ew((long)Bn * Pp * H), 256, 0, stream>>>(encP, encT);

    // ---- composer ----
    kproj_kernel<<<ew(Pp * H), 256, 0, stream>>>(prim_emb, Wk, bk, kbuf);
    gemm(stream, encT, 0, H, bWq, 0, bq, 0, H, nullptr, 0, 0, nullptr, 0, nullptr, 0, 0,
         qbuf, 0, Bn * Pp, H, 1, 0);
    gemm(stream, encT, 0, H, bWv, 0, bv, 0, H, nullptr, 0, 0, nullptr, 0, nullptr, 0, 0,
         vbuf, 0, Bn * Pp, H, 1, 0);
    attention_kernel<<<ew(Bn * NHh * Pp), 256, 0, stream>>>(qbuf, kbuf, vbuf, attb);
    gemm(stream, attb, 0, H, bWo, 0, bo, 0, H, nullptr, 0, 0, nullptr, 0, nullptr, 0, 0,
         attO, 0, Bn * Pp, H, 1, 0);
    // composed = relu(relu(cat @ Wc1^T + bc1) @ Wc2^T + bc2) + mean_p(attO)
    gemm(stream, encT, 0, Pp * H, bWc1, 0, bc1, 0, Pp * H,
         nullptr, 0, 0, nullptr, 0, nullptr, 0, 0,
         comp1, 0, Bn, 2 * H, 1, 1);
    gemm(stream, comp1, 0, 2 * H, bWc2, 0, bc2, 0, 2 * H,
         nullptr, 0, 0, nullptr, 0, nullptr, 0, 0,
         composed, 0, Bn, H, 1, 1);
    agg_add_kernel<<<ew((long)Bn * H), 256, 0, stream>>>(attO, composed);

    // ---- time embedding ----
    time_emb_kernel<<<ew(TF * H), 256, 0, stream>>>(Wt1, bt1, Wt2, bt2, te);

    // ---- decoder static parts (all M modes batched via grid.z) ----
    gemm(stream, composed, 0, H, bWm1, 2L * H * H, bm1, 2L * H, H,
         nullptr, 0, 0, nullptr, 0, nullptr, 0, 0,
         mf1, (long)Bn * 2 * H, Bn, 2 * H, Mm, 1);
    gemm(stream, mf1, (long)Bn * 2 * H, 2 * H, bWm2, 2L * H * H, bm2, (long)H, 2 * H,
         nullptr, 0, 0, nullptr, 0, nullptr, 0, 0,
         mf2b, (long)Bn * H, Bn, H, Mm, 1);
    gemm(stream, mf2b, (long)Bn * H, H, bWm3, (long)H * H, bm3, (long)H, H,
         nullptr, 0, 0, nullptr, 0, nullptr, 0, 0,
         mfb, (long)Bn * H, Bn, H, Mm, 0);
    gemm(stream, composed, 0, H, bWinit, 2L * H * H, binit, 2L * H, H,
         nullptr, 0, 0, nullptr, 0, nullptr, 0, 0,
         initb, (long)Bn * 2 * H, Bn, 2 * H, Mm, 0);
    split_init_kernel<<<ew((long)Mm * Bn * H), 256, 0, stream>>>(initb, dh0, dc0, dh1, dc1);

    // ---- decoder recurrence ----
    for (int t = 0; t < TF; ++t) {
        xin_kernel<<<ew((long)Mm * Bn * H), 256, 0, stream>>>(mfb, te, prevb, Wf, bf, xinb, t);
        gemm(stream,
             xinb, (long)Bn * H, H, bdWih0, 4L * H * H, dbih0, 4L * H, H,
             dh0,  (long)Bn * H, H, bdWhh0, 4L * H * H, dbhh0, 4L * H, H,
             dgates, (long)Bn * 4 * H, Bn, 4 * H, Mm, 0);
        lstm_cell_kernel<<<ew((long)Mm * Bn * H), 256, 0, stream>>>(dgates, dh0, dc0, Mm * Bn);
        gemm(stream,
             dh0, (long)Bn * H, H, bdWih1, 4L * H * H, dbih1, 4L * H, H,
             dh1, (long)Bn * H, H, bdWhh1, 4L * H * H, dbhh1, 4L * H, H,
             dgates, (long)Bn * 4 * H, Bn, 4 * H, Mm, 0);
        lstm_cell_kernel<<<ew((long)Mm * Bn * H), 256, 0, stream>>>(dgates, dh1, dc1, Mm * Bn);
        gemm(stream, dh1, (long)Bn * H, H, bWo1, (long)H * H, bo1, (long)H, H,
             nullptr, 0, 0, nullptr, 0, nullptr, 0, 0,
             o1b, (long)Bn * H, Bn, H, Mm, 1);
        out_head_kernel<<<ew(Mm * Bn), 256, 0, stream>>>(o1b, Wo2, bo2, prevb, outTrajs, t);
    }

    // ---- confidence head ----
    gemm(stream, composed, 0, H, bcW1, 0, cb1, 0, H,
         nullptr, 0, 0, nullptr, 0, nullptr, 0, 0,
         chid, 0, Bn, H / 2, 1, 1);
    conf_kernel<<<ew(Bn), 256, 0, stream>>>(chid, cW2, cb2, outConf, outLogits);
}
